// Model_18906446037699
// MI455X (gfx1250) — matrix operand order repaired and checked by reading, not compiled, not run on hardware
//
#include <hip/hip_runtime.h>
#include <hip/hip_bf16.h>

typedef __bf16 bf16;
typedef __attribute__((ext_vector_type(16))) __bf16 v16bf;
typedef __attribute__((ext_vector_type(8)))  float  v8f;
typedef unsigned int u32;
typedef __attribute__((ext_vector_type(4)))  u32    u32x4;
typedef int v4i __attribute__((vector_size(16)));    // matches builtin param type

union Frag16 { v16bf v; u32x4 q[2]; };

#define TM  128
#define TN  256
#define TK  32
#define LDT 40          // padded LDS row stride in bf16 elements (80B -> conflict-free b128)
#define STAT_CH 64      // row chunks for deterministic column stats

// ---- CDNA5 async global->LDS copy (ASYNCcnt path), with safe fallback ----
#if defined(__AMDGCN__) && __has_builtin(__builtin_amdgcn_global_load_async_to_lds_b128)
#define HAVE_ASYNC 1
#else
#define HAVE_ASYNC 0
#endif

#if HAVE_ASYNC
typedef __attribute__((address_space(1))) v4i gv4i;   // global (AS1) 16B vector
typedef __attribute__((address_space(3))) v4i lv4i;   // LDS    (AS3) 16B vector
__device__ __forceinline__ void copy16(const bf16* g, bf16* l) {
  __builtin_amdgcn_global_load_async_to_lds_b128((gv4i*)g, (lv4i*)l, 0, 0);
}
__device__ __forceinline__ void wait_async() {
#if __has_builtin(__builtin_amdgcn_s_wait_asynccnt)
  __builtin_amdgcn_s_wait_asynccnt(0);
#else
  asm volatile("s_wait_asynccnt 0x0" ::: "memory");
#endif
}
#else
__device__ __forceinline__ void copy16(const bf16* g, bf16* l) {
  *(u32x4*)l = *(const u32x4*)g;
}
__device__ __forceinline__ void wait_async() {}
#endif

// ---------------------------------------------------------------------------
// WMMA GEMM: C[M,1024] = A[M,Kp](bf16) x W[1024,Kp]^T(bf16) + bias, f32 out.
// Grid: (N/256, M/128), block 256 (8 waves, 2x4), per wave 64x64 out tile.
// Double-buffered LDS; tile k+1 async-DMA overlaps the 16 WMMAs of tile k.
// ---------------------------------------------------------------------------
__global__ __launch_bounds__(256)
void k_gemm_bias(const bf16* __restrict__ A, const bf16* __restrict__ W,
                 const float* __restrict__ bias, float* __restrict__ C,
                 int K, int ldc) {
  __shared__ bf16 As[2][TM * LDT];
  __shared__ bf16 Bs[2][TN * LDT];

  const int tid   = threadIdx.x;
  const int lane  = tid & 31;
  const int wave  = tid >> 5;
  const int waveM = wave >> 2;          // 0..1 (64 rows each)
  const int waveN = wave & 3;           // 0..3 (64 cols each)
  const int half  = lane >> 4;          // K-half selector per ISA layout
  const int m0    = blockIdx.y * TM;
  const int n0    = blockIdx.x * TN;

  v8f zero = {0.f,0.f,0.f,0.f,0.f,0.f,0.f,0.f};
  v8f acc[4][4];
  for (int mt = 0; mt < 4; ++mt)
    for (int nt = 0; nt < 4; ++nt) acc[mt][nt] = zero;

  const int srow = tid >> 2;            // 0..63
  const int scol = (tid & 3) * 8;       // 0,8,16,24 (8 bf16 = 16B)

  auto stage = [&](int bufi, int k0) {
    const bf16* a0 = A + (size_t)(m0 + srow) * K + k0 + scol;
    copy16(a0,                  &As[bufi][srow * LDT + scol]);
    copy16(a0 + (size_t)64 * K, &As[bufi][(srow + 64) * LDT + scol]);
    const bf16* b0 = W + (size_t)(n0 + srow) * K + k0 + scol;
    copy16(b0,                   &Bs[bufi][srow * LDT + scol]);
    copy16(b0 + (size_t)64  * K, &Bs[bufi][(srow + 64)  * LDT + scol]);
    copy16(b0 + (size_t)128 * K, &Bs[bufi][(srow + 128) * LDT + scol]);
    copy16(b0 + (size_t)192 * K, &Bs[bufi][(srow + 192) * LDT + scol]);
  };

  const int mrow = waveM * 64 + (lane & 15);
  const int nrow = waveN * 64 + (lane & 15);

  stage(0, 0);
  int buf = 0;
  for (int k0 = 0; k0 < K; k0 += TK) {
    wait_async();            // this buffer's DMA complete (per wave)
    __syncthreads();         // ... and visible across all waves
    if (k0 + TK < K) stage(buf ^ 1, k0 + TK);   // overlap next-tile DMA

    Frag16 afr[4];
    for (int mt = 0; mt < 4; ++mt) {
      const bf16* base = &As[buf][(mrow + mt * 16) * LDT];
      afr[mt].q[0] = *(const u32x4*)(base + half * 8);        // K 0..7 / 8..15
      afr[mt].q[1] = *(const u32x4*)(base + 16 + half * 8);   // K 16..23 / 24..31
    }
    for (int nt = 0; nt < 4; ++nt) {
      Frag16 bfr;
      const bf16* base = &Bs[buf][(nrow + nt * 16) * LDT]; // K 0..7,16..23 / 8..15,24..31
      bfr.q[0] = *(const u32x4*)(base + half * 8);
      bfr.q[1] = *(const u32x4*)(base + 16 + half * 8);
      for (int mt = 0; mt < 4; ++mt)
        acc[mt][nt] = __builtin_amdgcn_wmma_f32_16x16x32_bf16(
            false, afr[mt].v, false, bfr.v, (short)0, acc[mt][nt],
            false, false);
    }
    __syncthreads();         // done reading buf before it is re-staged
    buf ^= 1;
  }

  // Epilogue: bias add + f32 store. D layout: N=lane&15, M = v + 8*half.
  for (int mt = 0; mt < 4; ++mt)
    for (int nt = 0; nt < 4; ++nt) {
      const int n     = n0 + waveN * 64 + nt * 16 + (lane & 15);
      const int mbase = m0 + waveM * 64 + mt * 16 + half * 8;
      const float bb  = bias[n];
      for (int v = 0; v < 8; ++v)
        C[(size_t)(mbase + v) * ldc + n] = acc[mt][nt][v] + bb;
    }
}

// ---------------------------------------------------------------------------
// f32 [rows,K] -> bf16 [rows,Kp], zero padded (also used as plain converter).
// ---------------------------------------------------------------------------
__global__ void k_cvt_pad(const float* __restrict__ x, bf16* __restrict__ out,
                          int K, int Kp, size_t total) {
  size_t i = (size_t)blockIdx.x * blockDim.x + threadIdx.x;
  if (i >= total) return;
  size_t r = i / (size_t)Kp;
  int    c = (int)(i - r * (size_t)Kp);
  out[i] = (c < K) ? (bf16)x[r * (size_t)K + c] : (bf16)0.0f;
}

// Deterministic column stats, stage 1: partial sum/sumsq per row chunk.
__global__ void k_colstats1(const float* __restrict__ h, float* __restrict__ psum,
                            float* __restrict__ psq, int N, int rowsPer) {
  int c = blockIdx.x * 256 + threadIdx.x;
  if (c >= N) return;
  int r0 = blockIdx.y * rowsPer;
  float s = 0.f, q = 0.f;
  for (int r = r0; r < r0 + rowsPer; ++r) {
    float x = h[(size_t)r * N + c];
    s += x; q += x * x;
  }
  psum[(size_t)blockIdx.y * N + c] = s;
  psq [(size_t)blockIdx.y * N + c] = q;
}

// Stage 2: reduce chunks -> mean, rstd (biased var, eps=1e-3).
__global__ void k_colstats2(const float* __restrict__ psum, const float* __restrict__ psq,
                            int N, int nch, float Bf,
                            float* __restrict__ meanv, float* __restrict__ rstdv) {
  int c = blockIdx.x * 256 + threadIdx.x;
  if (c >= N) return;
  float s = 0.f, q = 0.f;
  for (int i = 0; i < nch; ++i) { s += psum[(size_t)i * N + c]; q += psq[(size_t)i * N + c]; }
  float m   = s / Bf;
  float var = q / Bf - m * m;
  meanv[c] = m;
  rstdv[c] = rsqrtf(var + 0.001f);
}

// Fused BN(batch stats) + ReLU + f32->bf16 convert.
__global__ void k_bnrelu_cvt(const float* __restrict__ h, const float* __restrict__ meanv,
                             const float* __restrict__ rstdv, const float* __restrict__ g,
                             const float* __restrict__ b, bf16* __restrict__ out,
                             int N, size_t total) {
  size_t i = (size_t)blockIdx.x * blockDim.x + threadIdx.x;
  if (i >= total) return;
  int c = (int)(i % (size_t)N);
  float y = (h[i] - meanv[c]) * rstdv[c] * g[c] + b[c];
  out[i] = (bf16)fmaxf(y, 0.0f);
}

// Tiny loc linear: [B,2] @ wloc^T -> [B,10] (K=2: plain VALU is optimal).
__global__ void k_loc(const float* __restrict__ lat, const float* __restrict__ lon,
                      const float* __restrict__ wloc, const float* __restrict__ bloc,
                      float* __restrict__ out, int B) {
  int r = blockIdx.x * 256 + threadIdx.x;
  if (r >= B) return;
  float a = lat[r], o = lon[r];
  for (int j = 0; j < 10; ++j)
    out[(size_t)r * 10 + j] = wloc[2 * j] * a + wloc[2 * j + 1] * o + bloc[j];
}

// Build [B,320] bf16: BN+ReLU(loc) cols 0..9 | tag cols 10..309 | zero pad.
__global__ void k_cat(const float* __restrict__ locraw, const float* __restrict__ meanv,
                      const float* __restrict__ rstdv, const float* __restrict__ g,
                      const float* __restrict__ b, const float* __restrict__ tag,
                      bf16* __restrict__ out, size_t total) {
  size_t i = (size_t)blockIdx.x * blockDim.x + threadIdx.x;
  if (i >= total) return;
  size_t r = i / 320;
  int    c = (int)(i - r * 320);
  if (c < 10) {
    float y = (locraw[r * 10 + c] - meanv[c]) * rstdv[c] * g[c] + b[c];
    out[i] = (bf16)fmaxf(y, 0.0f);
  } else if (c < 310) {
    out[i] = (bf16)tag[r * 300 + (c - 10)];
  } else {
    out[i] = (bf16)0.0f;
  }
}

__global__ void k_zero_counter(int* c) {
  if (blockIdx.x == 0 && threadIdx.x == 0) *c = 0;
}

// One wave (32 lanes) per row: both pairwise distances + margin test.
__global__ void k_dist(const float* __restrict__ anchor, const float* __restrict__ ip,
                       const float* __restrict__ inn, int* __restrict__ counter, int B) {
  int row  = (int)((blockIdx.x * (size_t)blockDim.x + threadIdx.x) >> 5);
  int lane = threadIdx.x & 31;
  if (row >= B) return;
  size_t base = (size_t)row * 1024;
  float sp = 0.f, sn = 0.f;
  for (int c = lane; c < 1024; c += 32) {
    float a  = anchor[base + c];
    float dp = fabsf(a - ip[base + c] + 1e-6f);
    float dn = fabsf(a - inn[base + c] + 1e-6f);
    sp += dp * dp; sn += dn * dn;
  }
  for (int off = 16; off > 0; off >>= 1) {
    sp += __shfl_down(sp, off, 32);
    sn += __shfl_down(sn, off, 32);
  }
  if (lane == 0 && (sqrtf(sn) - sqrtf(sp) > 0.5f)) atomicAdd(counter, 1);
}

__global__ void k_write_count(const int* c, float* out) {
  if (blockIdx.x == 0 && threadIdx.x == 0) out[0] = (float)(*c);
}

// ---------------------------------------------------------------------------
extern "C" void kernel_launch(void* const* d_in, const int* in_sizes, int n_in,
                              void* d_out, int out_size, void* d_ws, size_t ws_size,
                              hipStream_t stream) {
  const float* img_p = (const float*)d_in[0];
  const float* img_n = (const float*)d_in[1];
  const float* tag   = (const float*)d_in[2];
  const float* lat   = (const float*)d_in[3];
  const float* lon   = (const float*)d_in[4];
  const float* wi1 = (const float*)d_in[5];  const float* bi1 = (const float*)d_in[6];
  const float* gi1 = (const float*)d_in[7];  const float* ti1 = (const float*)d_in[8];
  const float* wi2 = (const float*)d_in[9];  const float* bi2 = (const float*)d_in[10];
  const float* wi3 = (const float*)d_in[11]; const float* bi3 = (const float*)d_in[12];
  const float* wloc = (const float*)d_in[13]; const float* bloc = (const float*)d_in[14];
  const float* gloc = (const float*)d_in[15]; const float* tloc = (const float*)d_in[16];
  const float* wm1 = (const float*)d_in[17]; const float* bm1 = (const float*)d_in[18];
  const float* gm1 = (const float*)d_in[19]; const float* tm1 = (const float*)d_in[20];
  const float* wm2 = (const float*)d_in[21]; const float* bm2 = (const float*)d_in[22];
  const float* gm2 = (const float*)d_in[23]; const float* tm2 = (const float*)d_in[24];
  const float* wm3 = (const float*)d_in[25]; const float* bm3 = (const float*)d_in[26];
  const float* gm3 = (const float*)d_in[27]; const float* tm3 = (const float*)d_in[28];
  const float* wm4 = (const float*)d_in[29]; const float* bm4 = (const float*)d_in[30];

  const int B = in_sizes[0] / 300;     // 32768
  const int N = 1024;

  // ---- workspace carve-up (aligned to 256B) ----
  char* wsp = (char*)d_ws;
  auto carve = [&](size_t bytes) -> void* {
    void* p = (void*)wsp;
    wsp += (bytes + 255) & ~(size_t)255;
    return p;
  };
  bf16* Wi1 = (bf16*)carve((size_t)1024 * 320 * 2);
  bf16* Wi2 = (bf16*)carve((size_t)1024 * 1024 * 2);
  bf16* Wi3 = (bf16*)carve((size_t)1024 * 1024 * 2);
  bf16* Wm1 = (bf16*)carve((size_t)1024 * 320 * 2);
  bf16* Wm2 = (bf16*)carve((size_t)1024 * 1024 * 2);
  bf16* Wm3 = (bf16*)carve((size_t)1024 * 1024 * 2);
  bf16* Wm4 = (bf16*)carve((size_t)1024 * 1024 * 2);
  bf16* Xbf = (bf16*)carve((size_t)B * 320 * 2);   // padded 320-wide inputs
  bf16* Abf = (bf16*)carve((size_t)B * 1024 * 2);  // bf16 activations
  float* H  = (float*)carve((size_t)B * 1024 * 4); // f32 pre-BN activations
  float* locraw = (float*)carve((size_t)B * 10 * 4);
  float* psum = (float*)carve((size_t)STAT_CH * 1024 * 4);
  float* psq  = (float*)carve((size_t)STAT_CH * 1024 * 4);
  float* meanv = (float*)carve(1024 * 4);
  float* rstdv = (float*)carve(1024 * 4);
  int*   counter = (int*)carve(256);

  float* out_ip     = (float*)d_out;
  float* out_inn    = out_ip  + (size_t)B * N;
  float* out_anchor = out_inn + (size_t)B * N;
  float* out_count  = (float*)d_out + (out_size - 1);

  // ---- helpers ----
  auto cvt = [&](const float* src, bf16* dst, int K, int Kp, size_t rows) {
    size_t total = rows * (size_t)Kp;
    k_cvt_pad<<<dim3((unsigned)((total + 255) / 256)), 256, 0, stream>>>(src, dst, K, Kp, total);
  };
  auto gemm = [&](const bf16* A, const bf16* W, const float* bias, float* C, int Kp) {
    dim3 grid(N / TN, B / TM);
    k_gemm_bias<<<grid, 256, 0, stream>>>(A, W, bias, C, Kp, N);
  };
  auto colstats = [&](const float* h, int nc) {
    dim3 g1((nc + 255) / 256, STAT_CH);
    k_colstats1<<<g1, 256, 0, stream>>>(h, psum, psq, nc, B / STAT_CH);
    k_colstats2<<<dim3((nc + 255) / 256), 256, 0, stream>>>(psum, psq, nc, STAT_CH,
                                                            (float)B, meanv, rstdv);
  };
  auto bnrelu = [&](const float* h, const float* g, const float* b, bf16* o) {
    size_t total = (size_t)B * N;
    k_bnrelu_cvt<<<dim3((unsigned)((total + 255) / 256)), 256, 0, stream>>>(
        h, meanv, rstdv, g, b, o, N, total);
  };

  // ---- weights -> bf16 (padded K) ----
  cvt(wi1, Wi1, 300, 320, 1024);
  cvt(wi2, Wi2, 1024, 1024, 1024);
  cvt(wi3, Wi3, 1024, 1024, 1024);
  cvt(wm1, Wm1, 310, 320, 1024);
  cvt(wm2, Wm2, 1024, 1024, 1024);
  cvt(wm3, Wm3, 1024, 1024, 1024);
  cvt(wm4, Wm4, 1024, 1024, 1024);

  // ---- image branch (positive) ----
  cvt(img_p, Xbf, 300, 320, (size_t)B);
  gemm(Xbf, Wi1, bi1, H, 320);
  colstats(H, N);
  bnrelu(H, gi1, ti1, Abf);
  gemm(Abf, Wi2, bi2, H, 1024);
  cvt(H, Abf, 1024, 1024, (size_t)B);
  gemm(Abf, Wi3, bi3, out_ip, 1024);

  // ---- image branch (negative, own batch stats) ----
  cvt(img_n, Xbf, 300, 320, (size_t)B);
  gemm(Xbf, Wi1, bi1, H, 320);
  colstats(H, N);
  bnrelu(H, gi1, ti1, Abf);
  gemm(Abf, Wi2, bi2, H, 1024);
  cvt(H, Abf, 1024, 1024, (size_t)B);
  gemm(Abf, Wi3, bi3, out_inn, 1024);

  // ---- location + multimodal branch ----
  k_loc<<<dim3((B + 255) / 256), 256, 0, stream>>>(lat, lon, wloc, bloc, locraw, B);
  colstats(locraw, 10);
  {
    size_t total = (size_t)B * 320;
    k_cat<<<dim3((unsigned)((total + 255) / 256)), 256, 0, stream>>>(
        locraw, meanv, rstdv, gloc, tloc, tag, Xbf, total);
  }
  gemm(Xbf, Wm1, bm1, H, 320);
  colstats(H, N); bnrelu(H, gm1, tm1, Abf);
  gemm(Abf, Wm2, bm2, H, 1024);
  colstats(H, N); bnrelu(H, gm2, tm2, Abf);
  gemm(Abf, Wm3, bm3, H, 1024);
  colstats(H, N); bnrelu(H, gm3, tm3, Abf);
  gemm(Abf, Wm4, bm4, out_anchor, 1024);

  // ---- distances + margin count ----
  k_zero_counter<<<1, 1, 0, stream>>>(counter);
  k_dist<<<dim3((B * 32 + 255) / 256), 256, 0, stream>>>(out_anchor, out_ip, out_inn,
                                                         counter, B);
  k_write_count<<<1, 1, 0, stream>>>(counter, out_count);
}
